// TSumTDisc_91001767067933
// MI455X (gfx1250) — matrix operand order repaired and checked by reading, not compiled, not run on hardware
//
#include <hip/hip_runtime.h>
#include <hip/hip_bf16.h>
#include <stdint.h>

typedef __attribute__((ext_vector_type(16))) _Float16 v16h;
typedef __attribute__((ext_vector_type(8)))  float    v8f;

#define SLOPE 0.01f
#define HID   40
#define TILES 4      // 16-point tiles per wave -> 64 points per wave, 512 per block
#define PPB   512    // points per block (8 waves * TILES * 16)
#define SLOTS1 192   // LDS segment slots for emb kernel (5 feats)
#define SLOTS3 160   // LDS segment slots for out kernel (32 feats)

// slope < 1  =>  leaky_relu(x) == max(x, slope*x)   (v_mul + v_max, no cndmask)
__device__ __forceinline__ float lrelu(float v) { return fmaxf(v, v * SLOPE); }

__device__ __forceinline__ v8f wmma_f16(v16h a, v16h b, v8f c) {
    // D = A(16x32 f16) * B(32x16 f16) + C(16x16 f32)
    return __builtin_amdgcn_wmma_f32_16x16x32_f16(false, a, false, b, (short)0, c, false, false);
}

// ---------------------------------------------------------------------------
// Zero helper
// ---------------------------------------------------------------------------
__global__ void k_zero(float* p, int n) {
    int i = blockIdx.x * blockDim.x + threadIdx.x;
    if (i < n) p[i] = 0.0f;
}

// ---------------------------------------------------------------------------
// Pack weights into WMMA B-fragment layout.
// Frag layout per ISA: lane<16 : element i = W[K=k0+i+(i<8?0:8)][N=n0+lane]
//                      lane>=16: element i = W[K=k0+8+i+(i<8?0:8)][N=n0+lane-16]
// Out-of-range (K>=Kdim || N>=Ndim) = 0, so padded tiles contribute exactly 0.
// pack[fid*512 + lane*16 + i]
// ---------------------------------------------------------------------------
__global__ void k_pack(const float* We1, const float* We2,
                       const float* Wo1, const float* Wo2,
                       _Float16* pack) {
    int l = threadIdx.x;
    if (l >= 32) return;
    auto packFrag = [&](int fid, const float* W, int Kdim, int Ndim, int k0, int n0) {
        int n  = n0 + (l & 15);
        int kb = k0 + ((l >> 4) << 3);
        for (int i = 0; i < 16; ++i) {
            int k = kb + (i < 8 ? i : 8 + i);
            float v = (k < Kdim && n < Ndim) ? W[k * Ndim + n] : 0.0f;
            pack[(size_t)fid * 512 + (size_t)l * 16 + i] = (_Float16)v;
        }
    };
    // emb layer1: W_emb1 [3,40] -> N tiles 0..2
    packFrag(0, We1, 3, 40, 0, 0);
    packFrag(1, We1, 3, 40, 0, 16);
    packFrag(2, We1, 3, 40, 0, 32);
    // emb layer2: W_emb2 [40,5] -> K steps 0,32 ; single N tile
    packFrag(3, We2, HID, 5, 0, 0);
    packFrag(4, We2, HID, 5, 32, 0);
    // out layer1: W_out1 [9,40] -> N tiles 0..2
    packFrag(5, Wo1, 9, 40, 0, 0);
    packFrag(6, Wo1, 9, 40, 0, 16);
    packFrag(7, Wo1, 9, 40, 0, 32);
    // out layer2: W_out2 [40,32] -> (Kstep,Ntile): (0,0)(0,1)(32,0)(32,1)
    packFrag(8,  Wo2, HID, 32, 0, 0);
    packFrag(9,  Wo2, HID, 32, 0, 16);
    packFrag(10, Wo2, HID, 32, 32, 0);
    packFrag(11, Wo2, HID, 32, 32, 16);
}

// ---------------------------------------------------------------------------
// Kernel: emb FFN (3->40->5) per point + segment-sum into x_aggr.
// Block = 256 threads = 8 waves; each wave owns TILES*16 = 64 contiguous points.
// ---------------------------------------------------------------------------
__global__ void k_emb(const float* __restrict__ x, const int* __restrict__ batch,
                      const v16h* __restrict__ packs,
                      _Float16* __restrict__ xe, float* __restrict__ x_aggr,
                      int Npts) {
    __shared__ _Float16 hbuf[8][16][48];
    __shared__ float    slds[SLOTS1 * 5];
    __shared__ int      s_batch[PPB];

    const int tid = threadIdx.x;
    const int l   = tid & 31;
    const int w   = tid >> 5;
    const int m   = l & 15;
    const long long blockBase = (long long)blockIdx.x * PPB;

    for (int i = tid; i < SLOTS1 * 5; i += 256) slds[i] = 0.0f;
    for (int i = tid; i < PPB; i += 256) {
        long long idx = blockBase + i;
        if (idx >= Npts) idx = Npts - 1;
        s_batch[i] = batch[idx];
    }
    __syncthreads();
    const int gfirst = s_batch[0];
    const int glast  = s_batch[PPB - 1];

    // hoist weight fragments (stay in VGPRs across tiles)
    const v16h b0 = packs[0 * 32 + l];
    const v16h b1 = packs[1 * 32 + l];
    const v16h b2 = packs[2 * 32 + l];
    const v16h b3 = packs[3 * 32 + l];
    const v16h b4 = packs[4 * 32 + l];

    const int rbase = (l < 16) ? 0 : 8;
    const int koff  = (l < 16) ? 0 : 8;

    for (int t = 0; t < TILES; ++t) {
        const long long base = blockBase + (long long)w * (TILES * 16) + t * 16;
        const int wrow = w * (TILES * 16) + t * 16;     // index into s_batch
        long long p  = base + m;
        long long pc = p < Npts ? p : (long long)(Npts - 1);

        // ---- layer 1: A = x padded [16 x 32], K0..2 valid (lanes<16 only) ----
        v16h a = {};
        if (l < 16) {
            a[0] = (_Float16)x[pc * 3 + 0];
            a[1] = (_Float16)x[pc * 3 + 1];
            a[2] = (_Float16)x[pc * 3 + 2];
        }
        v8f c0 = {}, c1 = {}, c2 = {};
        c0 = wmma_f16(a, b0, c0);
        c1 = wmma_f16(a, b1, c1);
        c2 = wmma_f16(a, b2, c2);

        // leaky + write transposed hidden to LDS: hbuf[row][feature]
        for (int r = 0; r < 8; ++r) {
            hbuf[w][rbase + r][m]      = (_Float16)lrelu(c0[r]);
            hbuf[w][rbase + r][m + 16] = (_Float16)lrelu(c1[r]);
            hbuf[w][rbase + r][m + 32] = (_Float16)lrelu(c2[r]);
        }
        __syncthreads();

        // ---- layer 2: K = 40 (two K-steps of 32), N tile 0 (5 valid) ----
        v16h a0, a1;
        for (int i = 0; i < 8; ++i) {
            a0[i]     = hbuf[w][m][koff + i];           // K 0..7  / 8..15
            a0[8 + i] = hbuf[w][m][koff + 16 + i];      // K 16..23 / 24..31
            a1[i]     = hbuf[w][m][koff + 32 + i];      // K 32..39 / 40..47 (==0)
            a1[8 + i] = (_Float16)0.0f;                 // K 48..63 pad
        }
        v8f d = {};
        d = wmma_f16(a0, b3, d);
        d = wmma_f16(a1, b4, d);

        // lane holds feature f=m (valid f<5) for 8 point-rows; leaky, store, reduce
        if (m < 5) {
            for (int r = 0; r < 8; ++r) {
                int rr = rbase + r;
                long long pp = base + rr;
                if (pp < Npts) {
                    float v = lrelu(d[r]);
                    xe[pp * 5 + m] = (_Float16)v;
                    int g = s_batch[wrow + rr];
                    int off = g - gfirst;
                    if (off >= 0 && off < SLOTS1)
                        atomicAdd(&slds[off * 5 + m], v);
                    else
                        atomicAdd(&x_aggr[(size_t)g * 5 + m], v);
                }
            }
        }
        __syncthreads();   // protect hbuf reuse next tile (uniform across block)
    }

    int nslots = glast - gfirst + 1;
    if (nslots > SLOTS1) nslots = SLOTS1;
    for (int i = tid; i < nslots * 5; i += 256) {
        float v = slds[i];
        if (v != 0.0f) atomicAdd(&x_aggr[(size_t)(gfirst + i / 5) * 5 + (i % 5)], v);
    }
}

// ---------------------------------------------------------------------------
// Kernel: global_nn (5->40->4) per graph. Tiny (12 MFLOP) — scalar.
// ---------------------------------------------------------------------------
__global__ void k_glob(const float* __restrict__ x_aggr,
                       const float* __restrict__ W1, const float* __restrict__ W2,
                       float* __restrict__ xglob, int B) {
    int g = blockIdx.x * blockDim.x + threadIdx.x;
    if (g >= B) return;
    float xin[5];
    for (int k = 0; k < 5; ++k) xin[k] = x_aggr[(size_t)g * 5 + k];
    float o[4] = {0.f, 0.f, 0.f, 0.f};
    for (int j = 0; j < HID; ++j) {
        float h = 0.f;
        for (int k = 0; k < 5; ++k) h += xin[k] * W1[k * HID + j];
        h = lrelu(h);
        for (int q = 0; q < 4; ++q) o[q] += h * W2[j * 4 + q];
    }
    for (int q = 0; q < 4; ++q) xglob[(size_t)g * 4 + q] = lrelu(o[q]);
}

// ---------------------------------------------------------------------------
// Kernel: out_nn (9->40->32) per point + segment-sum into pooled [B,32].
// Block = 256 threads = 8 waves; each wave owns 64 contiguous points.
// ---------------------------------------------------------------------------
__global__ void k_out(const _Float16* __restrict__ xe, const float* __restrict__ xglob,
                      const int* __restrict__ batch, const v16h* __restrict__ packs,
                      float* __restrict__ pooled, int Npts) {
    __shared__ _Float16 hbuf[8][16][48];
    __shared__ float    slds[SLOTS3 * 32];
    __shared__ int      s_batch[PPB];

    const int tid = threadIdx.x;
    const int l   = tid & 31;
    const int w   = tid >> 5;
    const int m   = l & 15;
    const long long blockBase = (long long)blockIdx.x * PPB;

    for (int i = tid; i < SLOTS3 * 32; i += 256) slds[i] = 0.0f;
    for (int i = tid; i < PPB; i += 256) {
        long long idx = blockBase + i;
        if (idx >= Npts) idx = Npts - 1;
        s_batch[i] = batch[idx];
    }
    __syncthreads();
    const int gfirst = s_batch[0];
    const int glast  = s_batch[PPB - 1];

    // hoist weight fragments
    const v16h b5  = packs[5  * 32 + l];
    const v16h b6  = packs[6  * 32 + l];
    const v16h b7  = packs[7  * 32 + l];
    const v16h b8  = packs[8  * 32 + l];
    const v16h b9  = packs[9  * 32 + l];
    const v16h b10 = packs[10 * 32 + l];
    const v16h b11 = packs[11 * 32 + l];

    const int rbase = (l < 16) ? 0 : 8;
    const int koff  = (l < 16) ? 0 : 8;

    for (int t = 0; t < TILES; ++t) {
        const long long base = blockBase + (long long)w * (TILES * 16) + t * 16;
        const int wrow = w * (TILES * 16) + t * 16;
        long long p  = base + m;
        long long pc = p < Npts ? p : (long long)(Npts - 1);
        const int g_m = s_batch[wrow + m];

        // ---- layer 1: A row = [xe(5) | xglob(4)] padded to K=32 ----
        v16h a = {};
        if (l < 16) {
            const _Float16* xr = &xe[pc * 5];
            a[0] = xr[0]; a[1] = xr[1]; a[2] = xr[2]; a[3] = xr[3]; a[4] = xr[4];
            const float* gr = &xglob[(size_t)g_m * 4];
            a[5] = (_Float16)gr[0]; a[6] = (_Float16)gr[1]; a[7] = (_Float16)gr[2];
        } else {
            a[0] = (_Float16)xglob[(size_t)g_m * 4 + 3];   // K = 8
        }
        v8f c0 = {}, c1 = {}, c2 = {};
        c0 = wmma_f16(a, b5, c0);
        c1 = wmma_f16(a, b6, c1);
        c2 = wmma_f16(a, b7, c2);

        for (int r = 0; r < 8; ++r) {
            hbuf[w][rbase + r][m]      = (_Float16)lrelu(c0[r]);
            hbuf[w][rbase + r][m + 16] = (_Float16)lrelu(c1[r]);
            hbuf[w][rbase + r][m + 32] = (_Float16)lrelu(c2[r]);
        }
        __syncthreads();

        // ---- layer 2: K=40 (two K-steps), N=32 (two tiles) ----
        v16h a0, a1;
        for (int i = 0; i < 8; ++i) {
            a0[i]     = hbuf[w][m][koff + i];
            a0[8 + i] = hbuf[w][m][koff + 16 + i];
            a1[i]     = hbuf[w][m][koff + 32 + i];       // cols 40..47 are exact 0
            a1[8 + i] = (_Float16)0.0f;
        }
        v8f d0 = {}, d1 = {};
        d0 = wmma_f16(a0, b8,  d0);
        d0 = wmma_f16(a1, b10, d0);
        d1 = wmma_f16(a0, b9,  d1);
        d1 = wmma_f16(a1, b11, d1);

        // leaky + segment-sum into pooled: lane holds features m and m+16
        for (int r = 0; r < 8; ++r) {
            int rr = rbase + r;
            long long pp = base + rr;
            if (pp < Npts) {
                float v0 = lrelu(d0[r]);
                float v1 = lrelu(d1[r]);
                int g = s_batch[wrow + rr];
                int off = g - gfirst;
                if (off >= 0 && off < SLOTS3) {
                    atomicAdd(&slds[off * 32 + m], v0);
                    atomicAdd(&slds[off * 32 + m + 16], v1);
                } else {
                    atomicAdd(&pooled[(size_t)g * 32 + m], v0);
                    atomicAdd(&pooled[(size_t)g * 32 + m + 16], v1);
                }
            }
        }
        __syncthreads();   // protect hbuf reuse next tile (uniform across block)
    }

    int nslots = glast - gfirst + 1;
    if (nslots > SLOTS3) nslots = SLOTS3;
    for (int i = tid; i < nslots * 32; i += 256) {
        float v = slds[i];
        if (v != 0.0f) atomicAdd(&pooled[(size_t)(gfirst + (i >> 5)) * 32 + (i & 31)], v);
    }
}

// ---------------------------------------------------------------------------
// Kernel: disc head (32->40->1), final linear. 43 MFLOP — scalar.
// ---------------------------------------------------------------------------
__global__ void k_disc(const float* __restrict__ pooled,
                       const float* __restrict__ W1, const float* __restrict__ W2,
                       float* __restrict__ out, int B) {
    int g = blockIdx.x * blockDim.x + threadIdx.x;
    if (g >= B) return;
    const float* pr = &pooled[(size_t)g * 32];
    float pl[32];
    for (int k = 0; k < 32; ++k) pl[k] = pr[k];
    float acc = 0.f;
    for (int j = 0; j < HID; ++j) {
        float h = 0.f;
        for (int k = 0; k < 32; ++k) h += pl[k] * W1[k * HID + j];
        acc += lrelu(h) * W2[j];
    }
    out[g] = acc;
}

// ---------------------------------------------------------------------------
// Launch
// ---------------------------------------------------------------------------
extern "C" void kernel_launch(void* const* d_in, const int* in_sizes, int n_in,
                              void* d_out, int out_size, void* d_ws, size_t ws_size,
                              hipStream_t stream) {
    const float* x      = (const float*)d_in[0];
    const int*   batch  = (const int*)d_in[1];
    const float* We1    = (const float*)d_in[2];
    const float* We2    = (const float*)d_in[3];
    const float* Wg1    = (const float*)d_in[4];
    const float* Wg2    = (const float*)d_in[5];
    const float* Wo1    = (const float*)d_in[6];
    const float* Wo2    = (const float*)d_in[7];
    const float* Wd1    = (const float*)d_in[8];
    const float* Wd2    = (const float*)d_in[9];
    float* out = (float*)d_out;

    const int Npts = in_sizes[0] / 3;
    const int B    = out_size;          // [B, 1] output

    // workspace carve-up (256B aligned)
    uintptr_t cur = (uintptr_t)d_ws;
    auto take = [&](size_t bytes) {
        cur = (cur + 255) & ~(uintptr_t)255;
        uintptr_t r = cur;
        cur += bytes;
        return (void*)r;
    };
    _Float16* xe     = (_Float16*)take((size_t)Npts * 5 * sizeof(_Float16));
    float*    x_aggr = (float*)   take((size_t)B * 5  * sizeof(float));
    float*    xglob  = (float*)   take((size_t)B * 4  * sizeof(float));
    float*    pooled = (float*)   take((size_t)B * 32 * sizeof(float));
    _Float16* pack   = (_Float16*)take((size_t)12 * 512 * sizeof(_Float16));
    const v16h* packs = (const v16h*)pack;

    // 1. zero accumulators + pack weight fragments
    k_zero<<<(B * 5  + 255) / 256, 256, 0, stream>>>(x_aggr, B * 5);
    k_zero<<<(B * 32 + 255) / 256, 256, 0, stream>>>(pooled, B * 32);
    k_pack<<<1, 32, 0, stream>>>(We1, We2, Wo1, Wo2, pack);

    // 2. emb FFN + segment sum
    int nBlocks = (Npts + PPB - 1) / PPB;
    k_emb<<<nBlocks, 256, 0, stream>>>(x, batch, packs, xe, x_aggr, Npts);

    // 3. global_nn per graph
    k_glob<<<(B + 255) / 256, 256, 0, stream>>>(x_aggr, Wg1, Wg2, xglob, B);

    // 4. out_nn + pooled segment sum (bulk of FLOPs, WMMA)
    k_out<<<nBlocks, 256, 0, stream>>>(xe, xglob, batch, packs, pooled, Npts);

    // 5. disc head
    k_disc<<<(B + 255) / 256, 256, 0, stream>>>(pooled, Wd1, Wd2, out, B);
}
